// MoE_68143951118426
// MI455X (gfx1250) — matrix operand order repaired and checked by reading, not compiled, not run on hardware
//
#include <hip/hip_runtime.h>
#include <hip/hip_bf16.h>
#include <math.h>

// ---- Problem constants (from reference) ----
static constexpr int kB    = 2;
static constexpr int kSLEN = 2048;
static constexpr int kS    = kB * kSLEN;   // 4096 tokens
static constexpr int kD    = 2048;
static constexpr int kE    = 8;
static constexpr int kF    = 5504;
static constexpr int kCap  = 1024;         // 2*S/E * 1.0
static constexpr float kEps = 1e-9f;

// ---- WMMA vector types (gfx1250, wave32) ----
typedef __bf16 bf16_t;
typedef bf16_t v16bf __attribute__((ext_vector_type(16)));
typedef float  v8f   __attribute__((ext_vector_type(8)));

__device__ __forceinline__ unsigned short f32_to_bf16(float f) {
  unsigned u = __builtin_bit_cast(unsigned, f);
  u += 0x7fffu + ((u >> 16) & 1u);          // round-to-nearest-even
  return (unsigned short)(u >> 16);
}
__device__ __forceinline__ unsigned pack_bf16x2(float lo, float hi) {
  return (unsigned)f32_to_bf16(lo) | ((unsigned)f32_to_bf16(hi) << 16);
}

union Frag32B { uint4 q[2]; v16bf v; };

// gfx1250 async global->LDS copy (ASYNCcnt tracked; no VGPR staging).
// %0 = per-lane LDS byte offset (VDST), %1 = per-lane 64-bit global address.
__device__ __forceinline__ void async_load_b128(unsigned lds_off, const void* gptr) {
  asm volatile("global_load_async_to_lds_b128 %0, %1, off"
               :: "v"(lds_off), "v"(gptr) : "memory");
}
__device__ __forceinline__ void wait_async0() {
  asm volatile("s_wait_asynccnt 0x0" ::: "memory");
}

// ---- GEMM tiling ----
static constexpr int BM = 128;
static constexpr int BN = 64;
static constexpr int BK = 32;
static constexpr int LDA_S = BK + 8;   // padded LDS strides (bf16 elems)
static constexpr int LDB_S = BK + 8;

// A fragment (16x32 bf16, MxK): lane<16 -> row lane, K {0..7,16..23};
// lane>=16 -> row lane-16, K {8..15,24..31}.
__device__ __forceinline__ v16bf frag_A(const unsigned short* rowbase, int lane) {
  Frag32B f;
  const int kb = (lane >= 16) ? 8 : 0;
  f.q[0] = *(const uint4*)(rowbase + kb);
  f.q[1] = *(const uint4*)(rowbase + 16 + kb);
  return f.v;
}

// B fragment (32x16 bf16, KxN) from LDS stored transposed [N][K]:
// lane<16 -> col lane, K {0..7,16..23} ; lane>=16 -> col lane-16, K {8..15,24..31}.
__device__ __forceinline__ v16bf frag_B(const unsigned short* colbase, int lane) {
  Frag32B f;
  const int kb = (lane >= 16) ? 8 : 0;
  f.q[0] = *(const uint4*)(colbase + kb);
  f.q[1] = *(const uint4*)(colbase + 16 + kb);
  return f.v;
}

// ---------------------------------------------------------------------------
// Kernel 1: zero xe (bf16 [E][Cap][D]) and gating accumulators
// ---------------------------------------------------------------------------
__global__ __launch_bounds__(256) void moe_zero(uint4* __restrict__ xe16,
                                                float* __restrict__ me_sum) {
  const size_t n16 = (size_t)kE * kCap * kD * 2 / 16;
  for (size_t i = (size_t)blockIdx.x * 256 + threadIdx.x; i < n16;
       i += (size_t)gridDim.x * 256) {
    uint4 z; z.x = 0; z.y = 0; z.z = 0; z.w = 0;
    xe16[i] = z;
  }
  if (blockIdx.x == 0 && threadIdx.x < kE) me_sum[threadIdx.x] = 0.f;
}

// ---------------------------------------------------------------------------
// Kernel 2: gating logits + softmax + top-2 (one token per thread)
// ---------------------------------------------------------------------------
__global__ __launch_bounds__(256)
void moe_gate_logits(const float* __restrict__ x, const float* __restrict__ wg,
                     int* __restrict__ tok_e1, int* __restrict__ tok_e2,
                     float* __restrict__ tok_g1, float* __restrict__ tok_g2,
                     float* __restrict__ me_sum) {
  const int s = blockIdx.x * blockDim.x + threadIdx.x;
  if (s >= kS) return;
  const float* xr = x + (size_t)s * kD;
  float acc[kE];
#pragma unroll
  for (int e = 0; e < kE; ++e) acc[e] = 0.f;
  for (int d = 0; d < kD; ++d) {
    const float xv = xr[d];
    const float4* wr = (const float4*)(wg + (size_t)d * kE);
    const float4 w0 = wr[0], w1v = wr[1];
    acc[0] += xv * w0.x;  acc[1] += xv * w0.y;
    acc[2] += xv * w0.z;  acc[3] += xv * w0.w;
    acc[4] += xv * w1v.x; acc[5] += xv * w1v.y;
    acc[6] += xv * w1v.z; acc[7] += xv * w1v.w;
  }
  float mx = acc[0];
#pragma unroll
  for (int e = 1; e < kE; ++e) mx = fmaxf(mx, acc[e]);
  float p[kE], sum = 0.f;
#pragma unroll
  for (int e = 0; e < kE; ++e) { p[e] = __expf(acc[e] - mx); sum += p[e]; }
  const float inv = 1.f / sum;
#pragma unroll
  for (int e = 0; e < kE; ++e) p[e] *= inv;
  int i1 = 0; float b1 = acc[0];
#pragma unroll
  for (int e = 1; e < kE; ++e) if (acc[e] > b1) { b1 = acc[e]; i1 = e; }
  int i2 = (i1 == 0) ? 1 : 0; float b2 = -3.4e38f;
#pragma unroll
  for (int e = 0; e < kE; ++e)
    if (e != i1 && acc[e] > b2) { b2 = acc[e]; i2 = e; }
  tok_e1[s] = i1; tok_e2[s] = i2;
  tok_g1[s] = p[i1]; tok_g2[s] = p[i2];
#pragma unroll
  for (int e = 0; e < kE; ++e) atomicAdd(&me_sum[e], p[e]);
}

// ---------------------------------------------------------------------------
// Kernel 3: capacity-slot assignment (single block; segmented prefix sums)
// Writes l_aux and exp_counts into the tail of d_out.
// ---------------------------------------------------------------------------
__global__ __launch_bounds__(256)
void moe_gate_assign(const int* __restrict__ tok_e1, const int* __restrict__ tok_e2,
                     const float* __restrict__ tok_g1, const float* __restrict__ tok_g2,
                     const float* __restrict__ me_sum,
                     int* __restrict__ tok_c1, int* __restrict__ tok_c2,
                     float* __restrict__ tok_w1, float* __restrict__ tok_w2,
                     float* __restrict__ out) {
  __shared__ int cnt1[256][kE];
  __shared__ int cnt2[256][kE];
  __shared__ int total1[kE];
  const int tid = threadIdx.x;
  const int per = kS / 256;          // 16 tokens per thread
  const int s0 = tid * per;
  int c1l[kE], c2l[kE];
#pragma unroll
  for (int e = 0; e < kE; ++e) { c1l[e] = 0; c2l[e] = 0; }
  for (int j = 0; j < per; ++j) {
    c1l[tok_e1[s0 + j]]++;
    c2l[tok_e2[s0 + j]]++;
  }
#pragma unroll
  for (int e = 0; e < kE; ++e) { cnt1[tid][e] = c1l[e]; cnt2[tid][e] = c2l[e]; }
  __syncthreads();
  if (tid < kE) {
    int run = 0;
    for (int t = 0; t < 256; ++t) { int v = cnt1[t][tid]; cnt1[t][tid] = run; run += v; }
    total1[tid] = run;
    run = 0;
    for (int t = 0; t < 256; ++t) { int v = cnt2[t][tid]; cnt2[t][tid] = run; run += v; }
  }
  __syncthreads();
  int r1[kE], r2[kE];
#pragma unroll
  for (int e = 0; e < kE; ++e) { r1[e] = cnt1[tid][e]; r2[e] = cnt2[tid][e]; }
  for (int j = 0; j < per; ++j) {
    const int s = s0 + j;
    const int e1 = tok_e1[s], e2 = tok_e2[s];
    int c1 = r1[e1]++;                     // position in cumsum(mask1)
    int c2 = total1[e2] + r2[e2]++;        // cumsum(mask2) + sum(mask1)
    float g1 = tok_g1[s], g2 = tok_g2[s];
    if (c1 >= kCap) { c1 = -1; g1 = 0.f; }
    if (c2 >= kCap) { c2 = -1; g2 = 0.f; }
    const float den = fmaxf(g1 + g2, kEps);
    tok_c1[s] = c1; tok_c2[s] = c2;
    tok_w1[s] = g1 / den; tok_w2[s] = g2 / den;
  }
  if (tid == 0) {
    // l_aux = mean_e(me*ce)*E^2 = E * sum_e(me*ce)
    float la = 0.f;
    const float invS = 1.f / (float)kS;
    for (int e = 0; e < kE; ++e)
      la += (me_sum[e] * invS) * ((float)total1[e] * invS);
    la *= (float)kE;
    out[(size_t)kS * kD] = la;
    int* ic = (int*)out + (size_t)kS * kD + 1;
    for (int e = 0; e < kE; ++e) ic[e] = total1[e];   // pre-capacity counts
  }
}

// ---------------------------------------------------------------------------
// Kernel 4: dispatch tokens into xe bf16 [E][Cap][D]
// ---------------------------------------------------------------------------
__global__ __launch_bounds__(256)
void moe_dispatch(const float* __restrict__ x,
                  const int* __restrict__ tok_e1, const int* __restrict__ tok_e2,
                  const int* __restrict__ tok_c1, const int* __restrict__ tok_c2,
                  unsigned short* __restrict__ xe) {
  const int s = blockIdx.x;
  const int which = blockIdx.y;
  const int e = which ? tok_e2[s] : tok_e1[s];
  const int c = which ? tok_c2[s] : tok_c1[s];
  if (c < 0) return;
  const float* src = x + (size_t)s * kD;
  unsigned short* dst = xe + ((size_t)e * kCap + c) * kD;
  const int i0 = threadIdx.x * 8;
  const float4 a = ((const float4*)(src + i0))[0];
  const float4 b = ((const float4*)(src + i0))[1];
  uint4 v;
  v.x = pack_bf16x2(a.x, a.y);
  v.y = pack_bf16x2(a.z, a.w);
  v.z = pack_bf16x2(b.x, b.y);
  v.w = pack_bf16x2(b.z, b.w);
  *(uint4*)(dst + i0) = v;
}

// ---------------------------------------------------------------------------
// Kernel 5: fused GEMM1 + SwiGLU.  H = silu(Xe@W1) * (Xe@W3), bf16 out.
// grid = (F/BN, Cap/BM, E), block = 256 (8 waves, 4x2 wave grid, 32x32/wave)
// Double-buffered LDS; A tile via GLOBAL_LOAD_ASYNC_TO_LDS_B128 overlapped
// with the WMMAs of the current tile; one barrier per K-step.
// ---------------------------------------------------------------------------
__global__ __launch_bounds__(256)
void moe_gemm1_swiglu(const unsigned short* __restrict__ xe,   // [E][Cap][D] bf16
                      const float* __restrict__ w1,            // [E][D][F] f32
                      const float* __restrict__ w3,
                      unsigned short* __restrict__ hout) {     // [E][Cap][F] bf16
  __shared__ unsigned short sA[2][BM * LDA_S];
  __shared__ unsigned short sB1[2][BN * LDB_S];
  __shared__ unsigned short sB3[2][BN * LDB_S];
  const int e  = blockIdx.z;
  const int m0 = blockIdx.y * BM;
  const int n0 = blockIdx.x * BN;
  const unsigned short* A = xe + (size_t)e * kCap * kD;
  const float* B1 = w1 + (size_t)e * kD * kF;
  const float* B3 = w3 + (size_t)e * kD * kF;
  unsigned short* Ho = hout + (size_t)e * kCap * kF;
  const int tid  = threadIdx.x;
  const int lane = tid & 31;
  const int wv   = tid >> 5;
  const int wm   = wv >> 1;   // 0..3
  const int wn   = wv & 1;    // 0..1
  const int l15  = lane & 15;

  v8f acc1[2][2], acc3[2][2];
#pragma unroll
  for (int mt = 0; mt < 2; ++mt)
#pragma unroll
    for (int nt = 0; nt < 2; ++nt)
#pragma unroll
      for (int i = 0; i < 8; ++i) { acc1[mt][nt][i] = 0.f; acc3[mt][nt][i] = 0.f; }

  // A-tile async destinations (per-lane LDS byte offsets; addr[31:0] of generic)
  const int arow0 = tid >> 2;                          // 0..63
  const int arow1 = arow0 + 64;                        // 64..127
  const int akc   = (tid & 3) << 3;                    // 0,8,16,24
  const unsigned sA0 = (unsigned)(size_t)&sA[0][0];
  const unsigned sA1 = (unsigned)(size_t)&sA[1][0];
  const unsigned aoff0 = (unsigned)((arow0 * LDA_S + akc) * 2);
  const unsigned aoff1 = (unsigned)((arow1 * LDA_S + akc) * 2);

  // B loader: each thread covers 2 K-rows x 4 N-cols, packs K-pairs into u32
  const int kk = (tid >> 4) << 1;   // 0,2,...,30
  const int nb = (tid & 15) << 2;   // 0..60

  float4 r1a, r1b, r3a, r3b;        // next-tile B staging registers

  auto issueA = [&](int buf, int k0) {
    const unsigned base = buf ? sA1 : sA0;
    async_load_b128(base + aoff0, A + (size_t)(m0 + arow0) * kD + k0 + akc);
    async_load_b128(base + aoff1, A + (size_t)(m0 + arow1) * kD + k0 + akc);
  };
  auto loadB = [&](int k0) {
    const float* p1 = B1 + (size_t)(k0 + kk) * kF + n0 + nb;
    const float* p3 = B3 + (size_t)(k0 + kk) * kF + n0 + nb;
    r1a = *(const float4*)p1; r1b = *(const float4*)(p1 + kF);
    r3a = *(const float4*)p3; r3b = *(const float4*)(p3 + kF);
  };
  auto storeB = [&](int buf) {
    unsigned short* b1p = &sB1[buf][0];
    unsigned short* b3p = &sB3[buf][0];
    const float l1[4] = {r1a.x, r1a.y, r1a.z, r1a.w};
    const float h1[4] = {r1b.x, r1b.y, r1b.z, r1b.w};
    const float l3[4] = {r3a.x, r3a.y, r3a.z, r3a.w};
    const float h3[4] = {r3b.x, r3b.y, r3b.z, r3b.w};
#pragma unroll
    for (int j = 0; j < 4; ++j) {
      *(unsigned*)&b1p[(nb + j) * LDB_S + kk] = pack_bf16x2(l1[j], h1[j]);
      *(unsigned*)&b3p[(nb + j) * LDB_S + kk] = pack_bf16x2(l3[j], h3[j]);
    }
  };

  // Prologue: fill buffer 0
  issueA(0, 0);
  loadB(0);
  storeB(0);

  const int KT = kD / BK;
  for (int kt = 0; kt < KT; ++kt) {
    const int cur = kt & 1;
    const int nxt = cur ^ 1;
    wait_async0();        // A[cur] resident (next-tile asyncs not yet issued)
    __syncthreads();      // B[cur] stores drained + all waves done reading nxt
    const bool hasNext = (kt + 1 < KT);
    if (hasNext) {
      issueA(nxt, (kt + 1) * BK);   // DMA overlapped with WMMAs below
      loadB((kt + 1) * BK);         // global latency hidden behind compute
    }
    v16bf af[2], b1f[2], b3f[2];
#pragma unroll
    for (int mt = 0; mt < 2; ++mt)
      af[mt] = frag_A(&sA[cur][(wm * 32 + mt * 16 + l15) * LDA_S], lane);
#pragma unroll
    for (int nt = 0; nt < 2; ++nt) {
      b1f[nt] = frag_B(&sB1[cur][(wn * 32 + nt * 16 + l15) * LDB_S], lane);
      b3f[nt] = frag_B(&sB3[cur][(wn * 32 + nt * 16 + l15) * LDB_S], lane);
    }
#pragma unroll
    for (int mt = 0; mt < 2; ++mt)
#pragma unroll
      for (int nt = 0; nt < 2; ++nt) {
        acc1[mt][nt] = __builtin_amdgcn_wmma_f32_16x16x32_bf16(
            false, af[mt], false, b1f[nt], (short)0, acc1[mt][nt], false, false);
        acc3[mt][nt] = __builtin_amdgcn_wmma_f32_16x16x32_bf16(
            false, af[mt], false, b3f[nt], (short)0, acc3[mt][nt], false, false);
      }
    if (hasNext) storeB(nxt);
  }
  // Epilogue: SwiGLU + bf16 store.  C layout: VGPR i -> M = i (+8 for hi lanes)
  const int hi8 = (lane >= 16) ? 8 : 0;
#pragma unroll
  for (int mt = 0; mt < 2; ++mt)
#pragma unroll
    for (int nt = 0; nt < 2; ++nt) {
      const int col = n0 + wn * 32 + nt * 16 + l15;
#pragma unroll
      for (int i = 0; i < 8; ++i) {
        const int row = m0 + wm * 32 + mt * 16 + hi8 + i;
        const float g = acc1[mt][nt][i];
        const float h = (g / (1.f + __expf(-g))) * acc3[mt][nt][i];
        Ho[(size_t)row * kF + col] = f32_to_bf16(h);
      }
    }
}

// ---------------------------------------------------------------------------
// Kernel 6: GEMM2.  eo = H @ W2, f32 out.  grid = (D/BN, Cap/BM, E)
// Same double-buffered async pipeline.
// ---------------------------------------------------------------------------
__global__ __launch_bounds__(256)
void moe_gemm2(const unsigned short* __restrict__ hin,   // [E][Cap][F] bf16
               const float* __restrict__ w2,             // [E][F][D] f32
               float* __restrict__ eo) {                  // [E][Cap][D] f32
  __shared__ unsigned short sA[2][BM * LDA_S];
  __shared__ unsigned short sB[2][BN * LDB_S];
  const int e  = blockIdx.z;
  const int m0 = blockIdx.y * BM;
  const int n0 = blockIdx.x * BN;
  const unsigned short* A = hin + (size_t)e * kCap * kF;
  const float* Bw = w2 + (size_t)e * kF * kD;
  float* Eo = eo + (size_t)e * kCap * kD;
  const int tid  = threadIdx.x;
  const int lane = tid & 31;
  const int wv   = tid >> 5;
  const int wm   = wv >> 1;
  const int wn   = wv & 1;
  const int l15  = lane & 15;

  v8f acc[2][2];
#pragma unroll
  for (int mt = 0; mt < 2; ++mt)
#pragma unroll
    for (int nt = 0; nt < 2; ++nt)
#pragma unroll
      for (int i = 0; i < 8; ++i) acc[mt][nt][i] = 0.f;

  const int arow0 = tid >> 2;
  const int arow1 = arow0 + 64;
  const int akc   = (tid & 3) << 3;
  const unsigned sA0 = (unsigned)(size_t)&sA[0][0];
  const unsigned sA1 = (unsigned)(size_t)&sA[1][0];
  const unsigned aoff0 = (unsigned)((arow0 * LDA_S + akc) * 2);
  const unsigned aoff1 = (unsigned)((arow1 * LDA_S + akc) * 2);

  const int kk = (tid >> 4) << 1;
  const int nb = (tid & 15) << 2;

  float4 ra, rb;

  auto issueA = [&](int buf, int k0) {
    const unsigned base = buf ? sA1 : sA0;
    async_load_b128(base + aoff0, A + (size_t)(m0 + arow0) * kF + k0 + akc);
    async_load_b128(base + aoff1, A + (size_t)(m0 + arow1) * kF + k0 + akc);
  };
  auto loadB = [&](int k0) {
    const float* p = Bw + (size_t)(k0 + kk) * kD + n0 + nb;
    ra = *(const float4*)p; rb = *(const float4*)(p + kD);
  };
  auto storeB = [&](int buf) {
    unsigned short* bp = &sB[buf][0];
    const float lo[4] = {ra.x, ra.y, ra.z, ra.w};
    const float hi[4] = {rb.x, rb.y, rb.z, rb.w};
#pragma unroll
    for (int j = 0; j < 4; ++j)
      *(unsigned*)&bp[(nb + j) * LDB_S + kk] = pack_bf16x2(lo[j], hi[j]);
  };

  issueA(0, 0);
  loadB(0);
  storeB(0);

  const int KT = kF / BK;
  for (int kt = 0; kt < KT; ++kt) {
    const int cur = kt & 1;
    const int nxt = cur ^ 1;
    wait_async0();
    __syncthreads();
    const bool hasNext = (kt + 1 < KT);
    if (hasNext) {
      issueA(nxt, (kt + 1) * BK);
      loadB((kt + 1) * BK);
    }
    v16bf af[2], bfr[2];
#pragma unroll
    for (int mt = 0; mt < 2; ++mt)
      af[mt] = frag_A(&sA[cur][(wm * 32 + mt * 16 + l15) * LDA_S], lane);
#pragma unroll
    for (int nt = 0; nt < 2; ++nt)
      bfr[nt] = frag_B(&sB[cur][(wn * 32 + nt * 16 + l15) * LDB_S], lane);
#pragma unroll
    for (int mt = 0; mt < 2; ++mt)
#pragma unroll
      for (int nt = 0; nt < 2; ++nt)
        acc[mt][nt] = __builtin_amdgcn_wmma_f32_16x16x32_bf16(
            false, af[mt], false, bfr[nt], (short)0, acc[mt][nt], false, false);
    if (hasNext) storeB(nxt);
  }
  const int hi8 = (lane >= 16) ? 8 : 0;
#pragma unroll
  for (int mt = 0; mt < 2; ++mt)
#pragma unroll
    for (int nt = 0; nt < 2; ++nt) {
      const int col = n0 + wn * 32 + nt * 16 + l15;
#pragma unroll
      for (int i = 0; i < 8; ++i) {
        const int row = m0 + wm * 32 + mt * 16 + hi8 + i;
        Eo[(size_t)row * kD + col] = acc[mt][nt][i];
      }
    }
}

// ---------------------------------------------------------------------------
// Kernel 7: combine back to token order
// ---------------------------------------------------------------------------
__global__ __launch_bounds__(256)
void moe_combine(const float* __restrict__ eo,
                 const int* __restrict__ tok_e1, const int* __restrict__ tok_e2,
                 const int* __restrict__ tok_c1, const int* __restrict__ tok_c2,
                 const float* __restrict__ tok_w1, const float* __restrict__ tok_w2,
                 float* __restrict__ out) {
  const int s = blockIdx.x;
  const int e1 = tok_e1[s], c1 = tok_c1[s];
  const int e2 = tok_e2[s], c2 = tok_c2[s];
  const float g1 = tok_w1[s], g2 = tok_w2[s];
  const int i0 = threadIdx.x * 8;
  float r[8];
#pragma unroll
  for (int i = 0; i < 8; ++i) r[i] = 0.f;
  if (c1 >= 0) {
    const float* p = eo + ((size_t)e1 * kCap + c1) * kD + i0;
#pragma unroll
    for (int i = 0; i < 8; ++i) r[i] += g1 * p[i];
  }
  if (c2 >= 0) {
    const float* p = eo + ((size_t)e2 * kCap + c2) * kD + i0;
#pragma unroll
    for (int i = 0; i < 8; ++i) r[i] += g2 * p[i];
  }
  float* o = out + (size_t)s * kD + i0;
#pragma unroll
  for (int i = 0; i < 8; ++i) o[i] = r[i];
}

// ---------------------------------------------------------------------------
extern "C" void kernel_launch(void* const* d_in, const int* in_sizes, int n_in,
                              void* d_out, int out_size, void* d_ws, size_t ws_size,
                              hipStream_t stream) {
  const float* x  = (const float*)d_in[0];   // [B,SLEN,D]
  const float* wg = (const float*)d_in[1];   // [D,E]
  const float* w1 = (const float*)d_in[2];   // [E,D,F]
  const float* w3 = (const float*)d_in[3];   // [E,D,F]
  const float* w2 = (const float*)d_in[4];   // [E,F,D]
  float* out = (float*)d_out;                // [S*D] ++ l_aux ++ exp_counts

  // Workspace layout (~191 MB total)
  char* ws = (char*)d_ws;
  size_t off = 0;
  auto take = [&](size_t bytes) -> void* {
    void* p = ws + off;
    off = (off + bytes + 255) & ~(size_t)255;
    return p;
  };
  unsigned short* xe = (unsigned short*)take((size_t)kE * kCap * kD * 2); // 33.5 MB
  unsigned short* hb = (unsigned short*)take((size_t)kE * kCap * kF * 2); // 90.2 MB
  float* eo     = (float*)take((size_t)kE * kCap * kD * 4);               // 67.1 MB
  int*   tok_e1 = (int*)take((size_t)kS * 4);
  int*   tok_e2 = (int*)take((size_t)kS * 4);
  int*   tok_c1 = (int*)take((size_t)kS * 4);
  int*   tok_c2 = (int*)take((size_t)kS * 4);
  float* tok_g1 = (float*)take((size_t)kS * 4);
  float* tok_g2 = (float*)take((size_t)kS * 4);
  float* tok_w1 = (float*)take((size_t)kS * 4);
  float* tok_w2 = (float*)take((size_t)kS * 4);
  float* me_sum = (float*)take((size_t)kE * 4);
  (void)ws_size; (void)in_sizes; (void)n_in; (void)out_size;

  moe_zero<<<dim3(2048), 256, 0, stream>>>((uint4*)xe, me_sum);
  moe_gate_logits<<<dim3(kS / 256), 256, 0, stream>>>(
      x, wg, tok_e1, tok_e2, tok_g1, tok_g2, me_sum);
  moe_gate_assign<<<dim3(1), 256, 0, stream>>>(
      tok_e1, tok_e2, tok_g1, tok_g2, me_sum,
      tok_c1, tok_c2, tok_w1, tok_w2, out);
  moe_dispatch<<<dim3(kS, 2), 256, 0, stream>>>(
      x, tok_e1, tok_e2, tok_c1, tok_c2, xe);
  moe_gemm1_swiglu<<<dim3(kF / BN, kCap / BM, kE), 256, 0, stream>>>(
      xe, w1, w3, hb);
  moe_gemm2<<<dim3(kD / BN, kCap / BM, kE), 256, 0, stream>>>(hb, w2, eo);
  moe_combine<<<dim3(kS), 256, 0, stream>>>(
      eo, tok_e1, tok_e2, tok_c1, tok_c2, tok_w1, tok_w2, out);
}
